// NottinghamModel_41755672052146
// MI455X (gfx1250) — matrix operand order repaired and checked by reading, not compiled, not run on hardware
//
#include <hip/hip_runtime.h>
#include <hip/hip_bf16.h>
#include <math.h>

// ---------------------------------------------------------------------------
// Nottingham 2-layer LSTM + dual softmax, CDNA5 (gfx1250) WMMA implementation
//   T=512, B=256, D=66, H=512, melody range R=34
// Recurrent GEMMs run on v_wmma_f32_16x16x32_f16; A/B fragments are loaded
// straight from global (L2/WGP$-resident) with contiguous-k b128 loads,
// so the inner loop has no LDS staging and no barriers.
// ---------------------------------------------------------------------------

#define TT   512
#define BB   256
#define DD   66
#define HH   512
#define RMEL 34
#define K0PAD 96            // x-part of layer0 K, padded 66 -> 96 (mult of 32)
#define KP0  (K0PAD + HH)   // layer0 padded K = 608
#define KP1  (2 * HH)       // layer1 K = 1024

typedef _Float16 h16;
typedef __attribute__((ext_vector_type(8)))  _Float16 v8h;
typedef __attribute__((ext_vector_type(16))) _Float16 v16h;
typedef __attribute__((ext_vector_type(8)))  float    v8f;

__device__ __forceinline__ float sigmf(float x) { return 1.0f / (1.0f + expf(-x)); }

// ---------------------------------------------------------------------------
// Setup kernels: f32 -> f16 conversion / padding / transposition, state zeroing
// ---------------------------------------------------------------------------

// x [T,B,66] f32 -> Xh [T,B,96] f16 (zero padded)
__global__ void convert_x_kernel(const float* __restrict__ x, h16* __restrict__ Xh) {
    size_t idx = (size_t)blockIdx.x * blockDim.x + threadIdx.x;
    size_t n = (size_t)TT * BB * K0PAD;
    if (idx >= n) return;
    int k = (int)(idx % K0PAD);
    size_t tb = idx / K0PAD;
    Xh[idx] = (k < DD) ? (h16)x[tb * DD + k] : (h16)0.0f;
}

// W0 [578, 2048] f32 -> W0t [2048, 608] f16 transposed + K-padded
__global__ void convert_w0_kernel(const float* __restrict__ W0, h16* __restrict__ W0t) {
    size_t idx = (size_t)blockIdx.x * blockDim.x + threadIdx.x;
    size_t n = (size_t)(4 * HH) * KP0;
    if (idx >= n) return;
    int k = (int)(idx % KP0);
    int ncol = (int)(idx / KP0);
    int korig = (k < DD) ? k : ((k >= K0PAD) ? (k - (K0PAD - DD)) : -1);
    W0t[idx] = (korig >= 0) ? (h16)W0[(size_t)korig * (4 * HH) + ncol] : (h16)0.0f;
}

// W1 [1024, 2048] f32 -> W1t [2048, 1024] f16 transposed
__global__ void convert_w1_kernel(const float* __restrict__ W1, h16* __restrict__ W1t) {
    size_t idx = (size_t)blockIdx.x * blockDim.x + threadIdx.x;
    size_t n = (size_t)(4 * HH) * KP1;
    if (idx >= n) return;
    int k = (int)(idx % KP1);
    int ncol = (int)(idx / KP1);
    W1t[idx] = (h16)W1[(size_t)k * (4 * HH) + ncol];
}

__global__ void zero_f32_kernel(float* __restrict__ p, int n) {
    int i = blockIdx.x * blockDim.x + threadIdx.x;
    if (i < n) p[i] = 0.0f;
}
__global__ void zero_f16_kernel(h16* __restrict__ p, int n) {
    int i = blockIdx.x * blockDim.x + threadIdx.x;
    if (i < n) p[i] = (h16)0.0f;
}

// ---------------------------------------------------------------------------
// Per-lane fragment helpers (contiguous k = hl*16 + s in A and B, not the ISA table order):
//  A 16x32 f16: lane = (hl, ln); lane holds row M=ln; K runs
//   [hl*16 .. hl*16+7] and [hl*16+8 .. hl*16+15] -> two contiguous 16B loads.
//  B 32x16 f16: lane holds column N=ln; K = hl*16 .. hl*16+15 -> one 32B load
//   from the K-contiguous transposed weight buffer.
// ---------------------------------------------------------------------------
__device__ __forceinline__ v16h load_afrag(const h16* __restrict__ arow, int k0, int hl) {
    v8h alo = *(const v8h*)(arow + k0 + hl * 16);
    v8h ahi = *(const v8h*)(arow + k0 + hl * 16 + 8);
    v16h a;
    #pragma unroll
    for (int i = 0; i < 8; ++i) { a[i] = alo[i]; a[8 + i] = ahi[i]; }
    return a;
}

// Accumulate nch K-chunks of [16 x K] @ [K x 16] for all four gates.
__device__ __forceinline__ void wmma_phase(const h16* __restrict__ arow, int nch, int hl,
                                           const h16* __restrict__ wp0,
                                           const h16* __restrict__ wp1,
                                           const h16* __restrict__ wp2,
                                           const h16* __restrict__ wp3,
                                           v8f& ai, v8f& aj, v8f& af, v8f& ao) {
    for (int kc = 0; kc < nch; ++kc) {
        const int k0 = kc << 5;
        v16h a  = load_afrag(arow, k0, hl);
        v16h b0 = *(const v16h*)(wp0 + k0);
        v16h b1 = *(const v16h*)(wp1 + k0);
        v16h b2 = *(const v16h*)(wp2 + k0);
        v16h b3 = *(const v16h*)(wp3 + k0);
        ai = __builtin_amdgcn_wmma_f32_16x16x32_f16(false, a, false, b0, (short)0, ai, false, false);
        aj = __builtin_amdgcn_wmma_f32_16x16x32_f16(false, a, false, b1, (short)0, aj, false, false);
        af = __builtin_amdgcn_wmma_f32_16x16x32_f16(false, a, false, b2, (short)0, af, false, false);
        ao = __builtin_amdgcn_wmma_f32_16x16x32_f16(false, a, false, b3, (short)0, ao, false, false);
    }
}

// ---------------------------------------------------------------------------
// Fused LSTM step: z = [A0 | A1] @ Wt^T + b ; gate nonlinearities ; h,c update
//   A0: [B, K0] f16   A1: [B, K1] f16   Wt: [4H, K0+K1] f16 (transposed)
// Each wave owns a 16x16 (batch x hcol) tile and accumulates all 4 gates,
// so the LSTM cell update fuses into the GEMM epilogue with no cross-wave
// communication. Grid: (B/16, H/16/8) = (16, 4), block = 256 (8 waves).
// ---------------------------------------------------------------------------
__global__ __launch_bounds__(256)
void lstm_step_kernel(const h16* __restrict__ A0, int K0,
                      const h16* __restrict__ A1, int K1,
                      const h16* __restrict__ Wt,
                      const float* __restrict__ bias,
                      const float* __restrict__ c_prev,
                      float* __restrict__ c_next,
                      h16*  __restrict__ h_next_h,
                      float* __restrict__ h_next_f)   // may be null
{
    const int KP   = K0 + K1;
    const int tid  = threadIdx.x;
    const int wave = tid >> 5;
    const int lane = tid & 31;
    const int hl   = lane >> 4;     // half-wave selector
    const int ln   = lane & 15;
    const int rt   = blockIdx.x;                 // batch row tile (0..15)
    const int ct   = blockIdx.y * 8 + wave;      // hidden col tile (0..31)
    const int row0 = rt * 16;
    const int col  = ct * 16 + ln;

    // Per-lane A row bases (lane ln owns matrix row M=ln in both half-waves).
    const h16* a0row = A0 + (size_t)(row0 + ln) * K0;
    const h16* a1row = A1 + (size_t)(row0 + ln) * K1;

    // Per-lane, per-gate W column bases (lane ln owns matrix column N=ln).
    const h16* wp0 = Wt + (size_t)(0 * HH + col) * KP + hl * 16;
    const h16* wp1 = Wt + (size_t)(1 * HH + col) * KP + hl * 16;
    const h16* wp2 = Wt + (size_t)(2 * HH + col) * KP + hl * 16;
    const h16* wp3 = Wt + (size_t)(3 * HH + col) * KP + hl * 16;

    // Warm the first weight lines (global_prefetch_b8).
    __builtin_prefetch(wp0, 0, 1);
    __builtin_prefetch(wp1, 0, 1);
    __builtin_prefetch(wp2, 0, 1);
    __builtin_prefetch(wp3, 0, 1);

    v8f acc_i = {}; v8f acc_j = {}; v8f acc_f = {}; v8f acc_o = {};

    // Phase 1: A0 columns of K, Phase 2: A1 columns of K (both mult of 32).
    const int nch0 = K0 >> 5;
    const int nch1 = K1 >> 5;
    wmma_phase(a0row, nch0, hl, wp0, wp1, wp2, wp3, acc_i, acc_j, acc_f, acc_o);
    wmma_phase(a1row, nch1, hl, wp0 + K0, wp1 + K0, wp2 + K0, wp3 + K0,
               acc_i, acc_j, acc_f, acc_o);

    // --- fused LSTM cell epilogue (gate order i, j, f, o; forget bias 1.0) ---
    const float bi = bias[0 * HH + col];
    const float bj = bias[1 * HH + col];
    const float bf = bias[2 * HH + col];
    const float bo = bias[3 * HH + col];
    const bool wf = (h_next_f != nullptr);

    #pragma unroll
    for (int r = 0; r < 8; ++r) {
        int row = row0 + r + hl * 8;            // C/D layout: VGPR r -> M=r (+8 for hi lanes)
        size_t o = (size_t)row * HH + col;
        float zi = acc_i[r] + bi;
        float zj = acc_j[r] + bj;
        float zf = acc_f[r] + bf;
        float zo = acc_o[r] + bo;
        float cp = c_prev[o];
        float c2 = cp * sigmf(zf + 1.0f) + sigmf(zi) * tanhf(zj);
        float h2 = tanhf(c2) * sigmf(zo);
        c_next[o]   = c2;
        h_next_h[o] = (h16)h2;
        if (wf) h_next_f[o] = h2;
    }
}

// ---------------------------------------------------------------------------
// Output projection for one timestep: logits[B,66] = h1[B,512] @ outW + outb
// Grid: (1, B), block 128 (threads 0..65 active per row).
// ---------------------------------------------------------------------------
__global__ __launch_bounds__(128)
void proj_kernel(const float* __restrict__ h1, const float* __restrict__ outW,
                 const float* __restrict__ outb, float* __restrict__ logits)
{
    int d = threadIdx.x;
    int b = blockIdx.y;
    if (d >= DD) return;
    const float* hr = h1 + (size_t)b * HH;
    float acc = outb[d];
    #pragma unroll 8
    for (int k = 0; k < HH; ++k) acc += hr[k] * outW[(size_t)k * DD + d];
    logits[(size_t)b * DD + d] = acc;
}

// ---------------------------------------------------------------------------
// Dual softmax (melody [0,34) / harmony [34,66)) in-place + per-sample loss
// One block of 64 threads per (t,b) sample.
// ---------------------------------------------------------------------------
__global__ __launch_bounds__(64)
void softmax_loss_kernel(float* __restrict__ probs, const int* __restrict__ targets,
                         float* __restrict__ partial)
{
    __shared__ float sl[DD];
    __shared__ float red[4];   // mel_max, mel_sum, har_max, har_sum
    size_t s = blockIdx.x;
    float* L = probs + s * DD;
    int tid = threadIdx.x;
    for (int i = tid; i < DD; i += 64) sl[i] = L[i];
    __syncthreads();
    if (tid == 0) {
        float m = sl[0];
        for (int i = 1; i < RMEL; ++i) m = fmaxf(m, sl[i]);
        float sum = 0.0f;
        for (int i = 0; i < RMEL; ++i) sum += expf(sl[i] - m);
        red[0] = m; red[1] = sum;
    } else if (tid == 32) {
        float m = sl[RMEL];
        for (int i = RMEL + 1; i < DD; ++i) m = fmaxf(m, sl[i]);
        float sum = 0.0f;
        for (int i = RMEL; i < DD; ++i) sum += expf(sl[i] - m);
        red[2] = m; red[3] = sum;
    }
    __syncthreads();
    for (int i = tid; i < DD; i += 64) {
        bool mel = (i < RMEL);
        float m = mel ? red[0] : red[2];
        float su = mel ? red[1] : red[3];
        L[i] = expf(sl[i] - m) / su;
    }
    if (tid == 0) {
        int tmel = targets[2 * s];
        int thar = targets[2 * s + 1];
        float lp_mel = sl[tmel] - red[0] - logf(red[1]);
        float lp_har = sl[RMEL + thar] - red[2] - logf(red[3]);
        partial[s] = 0.5f * (-lp_mel) + 0.5f * (-lp_har);
    }
}

// Deterministic two-pass reduction of per-sample losses.
__global__ __launch_bounds__(256)
void reduce_loss_kernel(const float* __restrict__ partial, int n, float* __restrict__ out)
{
    __shared__ float red[256];
    int tid = threadIdx.x;
    float acc = 0.0f;
    for (int i = tid; i < n; i += 256) acc += partial[i];   // fixed per-thread order
    red[tid] = acc;
    __syncthreads();
    for (int s = 128; s > 0; s >>= 1) {
        if (tid < s) red[tid] += red[tid + s];
        __syncthreads();
    }
    if (tid == 0) out[0] = red[0] / ((float)TT * (float)BB);
}

// ---------------------------------------------------------------------------
// Host-side orchestration
// ---------------------------------------------------------------------------
extern "C" void kernel_launch(void* const* d_in, const int* in_sizes, int n_in,
                              void* d_out, int out_size, void* d_ws, size_t ws_size,
                              hipStream_t stream) {
    (void)in_sizes; (void)n_in; (void)out_size; (void)ws_size;
    const float* x    = (const float*)d_in[0];
    const float* W0   = (const float*)d_in[1];
    const float* b0   = (const float*)d_in[2];
    const float* W1   = (const float*)d_in[3];
    const float* b1   = (const float*)d_in[4];
    const float* outW = (const float*)d_in[5];
    const float* outb = (const float*)d_in[6];
    const int*   tgt  = (const int*)d_in[7];

    char* ws = (char*)d_ws;
    size_t off = 0;
    auto alloc = [&](size_t bytes) -> void* {
        void* p = ws + off;
        off = (off + bytes + 255) & ~(size_t)255;
        return p;
    };

    h16*   Xh     = (h16*)alloc((size_t)TT * BB * K0PAD * sizeof(h16));   // 25.2 MB
    h16*   W0t    = (h16*)alloc((size_t)4 * HH * KP0 * sizeof(h16));      //  2.5 MB
    h16*   W1t    = (h16*)alloc((size_t)4 * HH * KP1 * sizeof(h16));      //  4.2 MB
    float* C0[2]  = { (float*)alloc((size_t)BB * HH * sizeof(float)),
                      (float*)alloc((size_t)BB * HH * sizeof(float)) };
    float* C1[2]  = { (float*)alloc((size_t)BB * HH * sizeof(float)),
                      (float*)alloc((size_t)BB * HH * sizeof(float)) };
    h16*   H0h[2] = { (h16*)alloc((size_t)BB * HH * sizeof(h16)),
                      (h16*)alloc((size_t)BB * HH * sizeof(h16)) };
    h16*   H1h[2] = { (h16*)alloc((size_t)BB * HH * sizeof(h16)),
                      (h16*)alloc((size_t)BB * HH * sizeof(h16)) };
    float* H1f    = (float*)alloc((size_t)BB * HH * sizeof(float));
    float* part   = (float*)alloc((size_t)TT * BB * sizeof(float));

    // --- setup: f16 conversion + transposition + state init (re-run every call) ---
    {
        size_t n = (size_t)TT * BB * K0PAD;
        convert_x_kernel<<<dim3((unsigned)((n + 255) / 256)), 256, 0, stream>>>(x, Xh);
    }
    {
        size_t n = (size_t)4 * HH * KP0;
        convert_w0_kernel<<<dim3((unsigned)((n + 255) / 256)), 256, 0, stream>>>(W0, W0t);
    }
    {
        size_t n = (size_t)4 * HH * KP1;
        convert_w1_kernel<<<dim3((unsigned)((n + 255) / 256)), 256, 0, stream>>>(W1, W1t);
    }
    {
        int n = BB * HH;
        dim3 g((n + 255) / 256);
        zero_f32_kernel<<<g, 256, 0, stream>>>(C0[0], n);
        zero_f32_kernel<<<g, 256, 0, stream>>>(C1[0], n);
        zero_f16_kernel<<<g, 256, 0, stream>>>(H0h[0], n);
        zero_f16_kernel<<<g, 256, 0, stream>>>(H1h[0], n);
    }

    // --- recurrence: per-step WMMA GEMM + fused cell update, ping-pong state ---
    dim3 lgrid(BB / 16, (HH / 16) / 8);   // (16, 4)
    for (int t = 0; t < TT; ++t) {
        int p = t & 1, q = p ^ 1;
        // layer 0: A = [x_t(padded) | h0_prev]
        lstm_step_kernel<<<lgrid, 256, 0, stream>>>(
            Xh + (size_t)t * BB * K0PAD, K0PAD, H0h[p], HH,
            W0t, b0, C0[p], C0[q], H0h[q], nullptr);
        // layer 1: A = [h0_new | h1_prev]
        lstm_step_kernel<<<lgrid, 256, 0, stream>>>(
            H0h[q], HH, H1h[p], HH,
            W1t, b1, C1[p], C1[q], H1h[q], H1f);
        // projection for this step's output directly into d_out logits region
        proj_kernel<<<dim3(1, BB), 128, 0, stream>>>(
            H1f, outW, outb, (float*)d_out + (size_t)t * BB * DD);
    }

    // --- dual softmax in place + deterministic loss reduction ---
    softmax_loss_kernel<<<dim3(TT * BB), 64, 0, stream>>>((float*)d_out, tgt, part);
    reduce_loss_kernel<<<dim3(1), 256, 0, stream>>>(part, TT * BB,
                                                    (float*)d_out + (size_t)TT * BB * DD);
}
